// GCNModel_20349555048560
// MI455X (gfx1250) — compile-verified, matrix operand order checked by reading, not run on hardware
//
#include <hip/hip_runtime.h>
#include <hip/hip_bf16.h>

typedef __bf16 v16bf __attribute__((ext_vector_type(16)));
typedef float  v8f   __attribute__((ext_vector_type(8)));

#define N_NODES 100000
#define N_EDGES 3200000
#define F_IN    512
#define HIDDEN  16
#define F_OUT   40
#define MTILES  (N_NODES / 16)   // 6250 exactly, no remainder

// ---------------- utility kernels ----------------

__global__ __launch_bounds__(256) void k_fill(float* p, float v, int n) {
    int i = blockIdx.x * 256 + threadIdx.x;
    if (i < n) p[i] = v;
}

__global__ __launch_bounds__(256) void k_edge_deg(const long long* __restrict__ dst,
                                                  float* __restrict__ deg) {
    int e = blockIdx.x * 256 + threadIdx.x;
    if (e < N_EDGES) atomicAdd(&deg[(int)dst[e]], 1.0f);
}

__global__ __launch_bounds__(256) void k_rsqrt(float* d, int n) {
    int i = blockIdx.x * 256 + threadIdx.x;
    if (i < n) d[i] = rsqrtf(d[i]);   // deg >= 1 always (self-loop baked in)
}

// Pack W1 [512,16] (row-major, K x N) into B-fragment layout:
// lane = column N (lanes 16-31 mirror with kbase=8), half j -> k pattern of the
// 16-bit 16x32 operand. Flat index == tid by construction.
__global__ __launch_bounds__(256) void k_pack_w1(const float* __restrict__ W1,
                                                 __bf16* __restrict__ w1p) {
    int tid  = blockIdx.x * 256 + threadIdx.x;       // 16*32*16 = 8192 threads
    int j    = tid & 15;
    int lane = (tid >> 4) & 31;
    int kt   = tid >> 9;
    int n    = lane & 15;
    int kbase = (lane & 16) ? 8 : 0;
    int k    = kt * 32 + kbase + (j & 7) + ((j >> 3) << 4);
    w1p[tid] = (__bf16)W1[k * HIDDEN + n];
}

// Pack W2 [16,40] into 3 column tiles of 16, K padded 16->32, N padded 40->48.
__global__ __launch_bounds__(256) void k_pack_w2(const float* __restrict__ W2,
                                                 __bf16* __restrict__ w2p) {
    int tid = blockIdx.x * 256 + threadIdx.x;        // 3*32*16 = 1536 threads
    if (tid >= 3 * 512) return;
    int j    = tid & 15;
    int lane = (tid >> 4) & 31;
    int t    = tid >> 9;
    int n    = t * 16 + (lane & 15);
    int kbase = (lane & 16) ? 8 : 0;
    int k    = kbase + (j & 7) + ((j >> 3) << 4);
    float v  = (k < HIDDEN && n < F_OUT) ? W2[k * F_OUT + n] : 0.0f;
    w2p[tid] = (__bf16)v;
}

// ---------------- GEMM1: h1[N,16] = x[N,512] @ W1 (bf16 WMMA, f32 acc) ------

__global__ __launch_bounds__(256) void k_gemm1(const float* __restrict__ x,
                                               const __bf16* __restrict__ w1p,
                                               float* __restrict__ h1) {
    const int lane = threadIdx.x & 31;
    const int wave = threadIdx.x >> 5;
    const int tile = blockIdx.x * 8 + wave;      // wave-uniform guard -> EXEC all 1s
    if (tile >= MTILES) return;

    const int m     = tile * 16 + (lane & 15);   // A row for this lane
    const int kbase = (lane & 16) ? 8 : 0;

    v8f c = {};
    const float* xr = x + (size_t)m * F_IN + kbase;

    for (int kt = 0; kt < F_IN / 32; ++kt) {
        __builtin_prefetch(xr + 32, 0, 1);       // next K-tile of the streaming read
        float4 a0 = *(const float4*)(xr + 0);
        float4 a1 = *(const float4*)(xr + 4);
        float4 a2 = *(const float4*)(xr + 16);
        float4 a3 = *(const float4*)(xr + 20);
        v16bf a;
        a[0]  = (__bf16)a0.x; a[1]  = (__bf16)a0.y; a[2]  = (__bf16)a0.z; a[3]  = (__bf16)a0.w;
        a[4]  = (__bf16)a1.x; a[5]  = (__bf16)a1.y; a[6]  = (__bf16)a1.z; a[7]  = (__bf16)a1.w;
        a[8]  = (__bf16)a2.x; a[9]  = (__bf16)a2.y; a[10] = (__bf16)a2.z; a[11] = (__bf16)a2.w;
        a[12] = (__bf16)a3.x; a[13] = (__bf16)a3.y; a[14] = (__bf16)a3.z; a[15] = (__bf16)a3.w;

        v16bf b = *(const v16bf*)(w1p + ((size_t)kt * 32 + lane) * 16);
        c = __builtin_amdgcn_wmma_f32_16x16x32_bf16(false, a, false, b,
                                                    (short)0, c, false, false);
        xr += 32;
    }

    // C/D layout: VGPR v -> row v + ((lane>>4)<<3), col = lane&15
    float* out = h1 + (size_t)(tile * 16 + ((lane >> 4) << 3)) * HIDDEN + (lane & 15);
    #pragma unroll
    for (int v = 0; v < 8; ++v) out[v * HIDDEN] = c[v];
}

// ---------------- GEMM2: h2[N,40] = hin[N,16] @ W2 (K padded to 32) ---------

__global__ __launch_bounds__(256) void k_gemm2(const float* __restrict__ hin,
                                               const __bf16* __restrict__ w2p,
                                               float* __restrict__ h2) {
    const int lane = threadIdx.x & 31;
    const int wave = threadIdx.x >> 5;
    const int tile = blockIdx.x * 8 + wave;
    if (tile >= MTILES) return;

    const int m     = tile * 16 + (lane & 15);
    const int kbase = (lane & 16) ? 8 : 0;

    const float* hr = hin + (size_t)m * HIDDEN + kbase;
    float4 a0 = *(const float4*)(hr);
    float4 a1 = *(const float4*)(hr + 4);
    v16bf a = {};                                  // halves 8..15 are the K>=16 pad
    a[0] = (__bf16)a0.x; a[1] = (__bf16)a0.y; a[2] = (__bf16)a0.z; a[3] = (__bf16)a0.w;
    a[4] = (__bf16)a1.x; a[5] = (__bf16)a1.y; a[6] = (__bf16)a1.z; a[7] = (__bf16)a1.w;

    #pragma unroll
    for (int t = 0; t < 3; ++t) {
        v8f c = {};
        v16bf b = *(const v16bf*)(w2p + ((size_t)t * 32 + lane) * 16);
        c = __builtin_amdgcn_wmma_f32_16x16x32_bf16(false, a, false, b,
                                                    (short)0, c, false, false);
        const int col = t * 16 + (lane & 15);
        if (col < F_OUT) {                         // divergence only around stores
            float* out = h2 + (size_t)(tile * 16 + ((lane >> 4) << 3)) * F_OUT + col;
            #pragma unroll
            for (int v = 0; v < 8; ++v) out[v * F_OUT] = c[v];
        }
    }
}

// ---------------- edge aggregation (gather * norm -> atomic scatter) --------

__global__ __launch_bounds__(256) void k_edge_agg16(const long long* __restrict__ src,
                                                    const long long* __restrict__ dst,
                                                    const float* __restrict__ dinv,
                                                    const float* __restrict__ h,
                                                    float* __restrict__ acc) {
    int e = blockIdx.x * 256 + threadIdx.x;
    if (e >= N_EDGES) return;
    int s = (int)src[e], d = (int)dst[e];
    float nrm = dinv[s] * dinv[d];
    const float* hr = h + (size_t)s * HIDDEN;
    float* ar = acc + (size_t)d * HIDDEN;
    #pragma unroll
    for (int q = 0; q < 4; ++q) {
        float4 v = *(const float4*)(hr + q * 4);
        atomicAdd(ar + q * 4 + 0, v.x * nrm);
        atomicAdd(ar + q * 4 + 1, v.y * nrm);
        atomicAdd(ar + q * 4 + 2, v.z * nrm);
        atomicAdd(ar + q * 4 + 3, v.w * nrm);
    }
}

__global__ __launch_bounds__(256) void k_edge_agg40(const long long* __restrict__ src,
                                                    const long long* __restrict__ dst,
                                                    const float* __restrict__ dinv,
                                                    const float* __restrict__ h,
                                                    float* __restrict__ acc) {
    int e = blockIdx.x * 256 + threadIdx.x;
    if (e >= N_EDGES) return;
    int s = (int)src[e], d = (int)dst[e];
    float nrm = dinv[s] * dinv[d];
    const float* hr = h + (size_t)s * F_OUT;
    float* ar = acc + (size_t)d * F_OUT;
    #pragma unroll
    for (int q = 0; q < 10; ++q) {
        float4 v = *(const float4*)(hr + q * 4);
        atomicAdd(ar + q * 4 + 0, v.x * nrm);
        atomicAdd(ar + q * 4 + 1, v.y * nrm);
        atomicAdd(ar + q * 4 + 2, v.z * nrm);
        atomicAdd(ar + q * 4 + 3, v.w * nrm);
    }
}

// self-loop contribution + bias (+ relu for layer 1), fused
__global__ __launch_bounds__(256) void k_finalize1(const float* __restrict__ h1,
                                                   const float* __restrict__ dinv,
                                                   const float* __restrict__ b1,
                                                   float* __restrict__ acc) {
    int i = blockIdx.x * 256 + threadIdx.x;
    if (i >= N_NODES * HIDDEN) return;
    int n = i >> 4, f = i & 15;
    float di = dinv[n];
    float v = acc[i] + h1[i] * di * di + b1[f];
    acc[i] = fmaxf(v, 0.0f);
}

__global__ __launch_bounds__(256) void k_finalize2(const float* __restrict__ h2,
                                                   const float* __restrict__ dinv,
                                                   const float* __restrict__ b2,
                                                   float* __restrict__ out) {
    int i = blockIdx.x * 256 + threadIdx.x;
    if (i >= N_NODES * F_OUT) return;
    int n = i / F_OUT, f = i % F_OUT;
    float di = dinv[n];
    out[i] = out[i] + h2[i] * di * di + b2[f];
}

// ---------------- launch ----------------

extern "C" void kernel_launch(void* const* d_in, const int* in_sizes, int n_in,
                              void* d_out, int out_size, void* d_ws, size_t ws_size,
                              hipStream_t stream) {
    const float*     x   = (const float*)d_in[0];
    const long long* ei  = (const long long*)d_in[1];   // int64 per reference
    const float*     W1  = (const float*)d_in[2];
    const float*     b1  = (const float*)d_in[3];
    const float*     W2  = (const float*)d_in[4];
    const float*     b2  = (const float*)d_in[5];
    const long long* src = ei;
    const long long* dst = ei + N_EDGES;
    float* out = (float*)d_out;

    // bump-allocate workspace (~29 MB total)
    char* ws = (char*)d_ws;
    size_t off = 0;
    auto walloc = [&](size_t bytes) -> void* {
        void* p = ws + off;
        off += (bytes + 255) & ~(size_t)255;
        return p;
    };
    float*  dinv = (float*) walloc((size_t)N_NODES * 4);
    __bf16* w1p  = (__bf16*)walloc((size_t)16 * 32 * 16 * 2);
    __bf16* w2p  = (__bf16*)walloc((size_t)3 * 32 * 16 * 2);
    float*  h1   = (float*) walloc((size_t)N_NODES * HIDDEN * 4);
    float*  acc1 = (float*) walloc((size_t)N_NODES * HIDDEN * 4);
    float*  h2   = (float*) walloc((size_t)N_NODES * F_OUT * 4);
    (void)in_sizes; (void)n_in; (void)out_size; (void)ws_size;

    int nb;
    // degrees (self-loop => init 1.0) and dinv = rsqrt(deg)
    nb = (N_NODES + 255) / 256;
    k_fill<<<nb, 256, 0, stream>>>(dinv, 1.0f, N_NODES);
    nb = (N_EDGES + 255) / 256;
    k_edge_deg<<<nb, 256, 0, stream>>>(dst, dinv);
    nb = (N_NODES + 255) / 256;
    k_rsqrt<<<nb, 256, 0, stream>>>(dinv, N_NODES);

    // pack weights into WMMA B-fragment layout (bf16)
    k_pack_w1<<<32, 256, 0, stream>>>(W1, w1p);
    k_pack_w2<<<6, 256, 0, stream>>>(W2, w2p);

    // zero accumulators (d_out doubles as layer-2 accumulator)
    nb = (N_NODES * HIDDEN + 255) / 256;
    k_fill<<<nb, 256, 0, stream>>>(acc1, 0.0f, N_NODES * HIDDEN);
    nb = (N_NODES * F_OUT + 255) / 256;
    k_fill<<<nb, 256, 0, stream>>>(out, 0.0f, N_NODES * F_OUT);

    // layer 1
    k_gemm1<<<(MTILES + 7) / 8, 256, 0, stream>>>(x, w1p, h1);
    nb = (N_EDGES + 255) / 256;
    k_edge_agg16<<<nb, 256, 0, stream>>>(src, dst, dinv, h1, acc1);
    nb = (N_NODES * HIDDEN + 255) / 256;
    k_finalize1<<<nb, 256, 0, stream>>>(h1, dinv, b1, acc1);

    // layer 2
    k_gemm2<<<(MTILES + 7) / 8, 256, 0, stream>>>(acc1, w2p, h2);
    nb = (N_EDGES + 255) / 256;
    k_edge_agg40<<<nb, 256, 0, stream>>>(src, dst, dinv, h2, out);
    nb = (N_NODES * F_OUT + 255) / 256;
    k_finalize2<<<nb, 256, 0, stream>>>(h2, dinv, b2, out);
}
